// TemplatePairEmbedder_66520453481103
// MI455X (gfx1250) — matrix operand order repaired and checked by reading, not compiled, not run on hardware
//
#include <hip/hip_runtime.h>

// ---------------------------------------------------------------------------
// TemplatePairEmbedder as a multi-hot K=96 GEMM on CDNA5 WMMA (gfx1250).
//
// B=2, T=4, L=384, C_Z=128, IN_FEAT=82 (39 dist bins + mask + 21 + 21).
// out[b,i,j,:] = (feat[b,i,j,:] @ W_ext) * res_pair / max(nvalid,1)
// where feat is a small-integer (0..4) multi-hot count vector over 82
// features (bias+mask merged into feature 39), padded to K=96.
// Counts are exact in f16; weights rounded to f16 (~1e-3 rel).
// Output write (151 MB fp32) bounds the kernel at ~6.5us on 23.3 TB/s HBM.
//
// Pipeline: double-buffered A tile; 16 builder lanes (2 per wave) construct
// tile jt+1 serially (no atomics, no convert pass) while all 8 waves run
// 3x v_wmma_f32_16x16x32_f16 on tile jt. One s_barrier per tile.
// ---------------------------------------------------------------------------

typedef _Float16 half_t;
typedef __attribute__((ext_vector_type(16))) _Float16 v16h;
typedef __attribute__((ext_vector_type(8)))  _Float16 v8h;
typedef __attribute__((ext_vector_type(8)))  float    v8f;
typedef __attribute__((ext_vector_type(4)))  float    v4f;

#define B_   2
#define T_   4
#define L_   384
#define CZ   128
#define NF   82     // IN_FEAT
#define KP   96     // K padded to 3 x 32
#define KST  104    // LDS row stride (halves); 52 dwords -> conflict-free frags
#define NBINS 39
#define NTILE (L_ / 16)

// --- prep: W_ext (f16, [CZ][KP], bias folded into feature 39) into d_ws ----
__global__ void prep_wt(const float* __restrict__ w, const float* __restrict__ bias,
                        half_t* __restrict__ wt) {
    int idx = blockIdx.x * blockDim.x + threadIdx.x;     // over CZ*KP
    if (idx >= CZ * KP) return;
    int c = idx / KP, k = idx % KP;
    float v = 0.0f;
    if (k < NF) {
        v = w[k * CZ + c];
        if (k == NBINS) v += bias[c];                    // merge bias with W_m row
    }
    wt[c * KP + k] = (half_t)v;
}

// --------------------------- main kernel -----------------------------------
__launch_bounds__(256)
__global__ void tpe_wmma_kernel(const int* __restrict__ aatype,
                                const float* __restrict__ coords,
                                const unsigned char* __restrict__ tmask,
                                const unsigned char* __restrict__ rmask,
                                const half_t* __restrict__ wt_g,
                                float* __restrict__ out) {
    __shared__ __align__(16) half_t sWT[CZ * KST];       // weights, f16, padded
    __shared__ __align__(16) half_t sA[2][16 * KST];     // double-buffered A tile
    __shared__ __align__(16) float  sC[T_ * L_ * 3];     // coords for batch b
    __shared__ int           sAat[T_ * L_];              // clipped aatype
    __shared__ __align__(4) unsigned char sTM[T_ * L_];  // template mask
    __shared__ __align__(4) unsigned char sRM[L_];       // residue mask
    __shared__ __align__(16) float sScl[2][16];          // per-row output scale

    const int tid = threadIdx.x;
    const int b   = blockIdx.x / L_;
    const int i   = blockIdx.x % L_;

    if (tid == 0) __builtin_prefetch(wt_g, 0, 3);        // global_prefetch_b8

    // ---- stage W_ext: global [CZ][KP] f16 -> LDS [CZ][KST] ----
    {
        const unsigned int* src = (const unsigned int*)wt_g;   // 2 halves / dword
        unsigned int* dst = (unsigned int*)sWT;
        for (int idx = tid; idx < CZ * (KP / 2); idx += 256) {
            int c = idx / (KP / 2), kk = idx % (KP / 2);
            dst[c * (KST / 2) + kk] = src[idx];
        }
        for (int idx = tid; idx < CZ * ((KST - KP) / 2); idx += 256) {
            int c = idx / ((KST - KP) / 2), kk = idx % ((KST - KP) / 2);
            dst[c * (KST / 2) + (KP / 2) + kk] = 0u;           // zero pad K=96..103
        }
    }
    // ---- stage per-batch coords / aatype / masks ----
    {
        const float* csrc = coords + (size_t)b * T_ * L_ * 3;
        for (int idx = tid; idx < T_ * L_ * 3; idx += 256) sC[idx] = csrc[idx];
        const int* asrc = aatype + b * T_ * L_;
        for (int idx = tid; idx < T_ * L_; idx += 256) {
            int a = asrc[idx];
            sAat[idx] = a < 0 ? 0 : (a > 20 ? 20 : a);
        }
        const unsigned int* msrc = (const unsigned int*)(tmask + (size_t)b * T_ * L_);
        unsigned int* mdst = (unsigned int*)sTM;
        for (int idx = tid; idx < (T_ * L_) / 4; idx += 256) mdst[idx] = msrc[idx];
        const unsigned int* rsrc = (const unsigned int*)(rmask + (size_t)b * L_);
        unsigned int* rdst = (unsigned int*)sRM;
        for (int idx = tid; idx < L_ / 4; idx += 256) rdst[idx] = rsrc[idx];
    }
    __syncthreads();

    // ---- B fragments: one 16-channel N-tile per wave, resident in VGPRs ----
    const int lane = tid & 31;
    const int ch0  = (tid >> 5) * 16;
    const int col  = ch0 + (lane & 15);
    union Frag { v16h v; v8h h[2]; };
    Frag bf[3];
#pragma unroll
    for (int ks = 0; ks < 3; ++ks) {
        // B 32x16 in this code's contiguous K order (same as A): lane holds one column; lanes 0-15 K=0..15, 16-31 K=16..31
        int k0 = ks * 32 + ((lane & 16) ? 16 : 0);
        const v8h* p = (const v8h*)(sWT + col * KST + k0);
        bf[ks].h[0] = p[0];
        bf[ks].h[1] = p[1];
    }

    // ---- builder-lane invariants (one builder per half-wave: tid%16==0) ----
    const bool isBuilder = (tid & 15) == 0;
    const int  brow      = tid >> 4;                     // 0..15
    float bcx[T_], bcy[T_], bcz[T_];
    int   bai[T_];
    unsigned char btmI[T_];
    if (isBuilder) {
#pragma unroll
        for (int t = 0; t < T_; ++t) {
            btmI[t] = sTM[t * L_ + i];
            bai[t]  = sAat[t * L_ + i];
            bcx[t]  = sC[(t * L_ + i) * 3 + 0];
            bcy[t]  = sC[(t * L_ + i) * 3 + 1];
            bcz[t]  = sC[(t * L_ + i) * 3 + 2];
        }
    }
    const float inv_step = 37.0f / 47.5f;                // 1 / linspace step
    const unsigned char rmi = sRM[i];
    float* outb = out + (((size_t)b * L_ + i) * L_) * CZ;

    // serial per-row tile builder: zero row, f16 "+= 1.0" RMW counts, scale
    auto build_tile = [&](int pb, int jt) {
        if (!isBuilder) return;
        const int j = jt * 16 + brow;
        half_t* arow = sA[pb] + brow * KST;
        uint4* az = (uint4*)arow;
        const uint4 z4 = {0u, 0u, 0u, 0u};
#pragma unroll
        for (int q = 0; q < KST / 8; ++q) az[q] = z4;    // 13 x ds_store_b128
        int nv = 0;
        const half_t one = (half_t)1.0f;
#pragma unroll
        for (int t = 0; t < T_; ++t) {
            if (btmI[t] & sTM[t * L_ + j]) {
                ++nv;
                float dx = bcx[t] - sC[(t * L_ + j) * 3 + 0];
                float dy = bcy[t] - sC[(t * L_ + j) * 3 + 1];
                float dz = bcz[t] - sC[(t * L_ + j) * 3 + 2];
                float d  = __builtin_sqrtf(dx * dx + dy * dy + dz * dz);
                int bin  = (int)__builtin_ceilf((d - 3.25f) * inv_step);
                bin = bin < 0 ? 0 : (bin > 38 ? 38 : bin);
                const int aj = sAat[t * L_ + j];
                arow[bin]          += one;               // W_d[bin]
                arow[NBINS]        += one;               // W_m + bias row
                arow[40 + bai[t]]  += one;               // W_i[aat_i]
                arow[61 + aj]      += one;               // W_j[aat_j]
            }
        }
        float denom = nv > 1 ? (float)nv : 1.0f;
        sScl[pb][brow] = (rmi & sRM[j]) ? (1.0f / denom) : 0.0f;
    };

    build_tile(0, 0);
    __syncthreads();

    const int row  = lane & 15;
    const int offA = (lane & 16) ? 16 : 0;               // contiguous K, matches B
    const int base = (lane & 16) ? 8 : 0;                // C/D: M = base + r

    for (int jt = 0; jt < NTILE; ++jt) {
        const int p  = jt & 1;
        const int j0 = jt * 16;

        if (jt + 1 < NTILE) build_tile(p ^ 1, jt + 1);   // overlap next build

        // 3 x v_wmma_f32_16x16x32_f16, chained f32 accumulator
        v8f c = {};
#pragma unroll
        for (int ks = 0; ks < 3; ++ks) {
            Frag af;
            const v8h* pa = (const v8h*)(sA[p] + row * KST + ks * 32 + offA);
            af.h[0] = pa[0];                             // K = k0    .. k0+7
            af.h[1] = pa[1];                             // K = k0+8  .. k0+15
            c = __builtin_amdgcn_wmma_f32_16x16x32_f16(
                    false, af.v, false, bf[ks].v, (short)0, c, false, false);
        }
        v4f s0 = *(const v4f*)(&sScl[p][base]);
        v4f s1 = *(const v4f*)(&sScl[p][base + 4]);
        float* op = outb + (size_t)(j0 + base) * CZ + ch0 + (lane & 15);
        op[0 * CZ] = c[0] * s0[0];
        op[1 * CZ] = c[1] * s0[1];
        op[2 * CZ] = c[2] * s0[2];
        op[3 * CZ] = c[3] * s0[3];
        op[4 * CZ] = c[4] * s1[0];
        op[5 * CZ] = c[5] * s1[1];
        op[6 * CZ] = c[6] * s1[2];
        op[7 * CZ] = c[7] * s1[3];

        __syncthreads();                                 // one barrier per tile
    }
}

extern "C" void kernel_launch(void* const* d_in, const int* in_sizes, int n_in,
                              void* d_out, int out_size, void* d_ws, size_t ws_size,
                              hipStream_t stream) {
    const int*           aatype = (const int*)d_in[0];           // (B,T,L) int32
    const float*         coords = (const float*)d_in[1];         // (B,T,L,3) f32
    const unsigned char* tmask  = (const unsigned char*)d_in[2]; // (B,T,L) bool
    const unsigned char* rmask  = (const unsigned char*)d_in[3]; // (B,L) bool
    const float*         weight = (const float*)d_in[4];         // (82,128) f32
    const float*         bias   = (const float*)d_in[5];         // (128,) f32

    half_t* wt = (half_t*)d_ws;                                  // 24 KB scratch

    prep_wt<<<(CZ * KP + 255) / 256, 256, 0, stream>>>(weight, bias, wt);
    tpe_wmma_kernel<<<B_ * L_, 256, 0, stream>>>(aatype, coords, tmask, rmask, wt,
                                                 (float*)d_out);
}
